// ChebNetII_65163243815286
// MI455X (gfx1250) — matrix operand order repaired and checked by reading, not compiled, not run on hardware
//
#include <hip/hip_runtime.h>

typedef __attribute__((ext_vector_type(16))) _Float16     v16h;
typedef __attribute__((ext_vector_type(8)))  float        v8f;
typedef __attribute__((ext_vector_type(4)))  unsigned int v4u;
typedef __attribute__((ext_vector_type(8)))  int          v8i_t;
typedef __attribute__((ext_vector_type(4)))  int          v4i_t;

#define NN   100000
#define EE   3200000
#define KORD 10
#define CIN  512
#define CH   64

#if defined(__has_builtin)
#  if __has_builtin(__builtin_amdgcn_tensor_load_to_lds) && \
      __has_builtin(__builtin_amdgcn_s_wait_tensorcnt)
#    define USE_TDM 1
#  endif
#endif
#ifndef USE_TDM
#  define USE_TDM 0
#endif

// ---------------------------------------------------------------- utilities
__global__ __launch_bounds__(256) void zero_f32(float* __restrict__ p, long long n) {
    long long i = (long long)blockIdx.x * blockDim.x + threadIdx.x;
    if (i < n) p[i] = 0.0f;
}

__global__ __launch_bounds__(256) void degree_kernel(const int* __restrict__ dst,
                                                     float* __restrict__ deg) {
    int e = blockIdx.x * blockDim.x + threadIdx.x;
    if (e < EE) atomicAdd(&deg[dst[e]], 1.0f);
}

__global__ __launch_bounds__(256) void dinv_kernel(const float* __restrict__ deg,
                                                   float* __restrict__ dinv) {
    int i = blockIdx.x * blockDim.x + threadIdx.x;
    if (i < NN) {
        float d = deg[i];
        dinv[i] = (d > 0.0f) ? rsqrtf(fmaxf(d, 1e-12f)) : 0.0f;
    }
}

__global__ __launch_bounds__(256) void norm_kernel(const int* __restrict__ src,
                                                   const int* __restrict__ dst,
                                                   const float* __restrict__ dinv,
                                                   float* __restrict__ nrm) {
    int e = blockIdx.x * blockDim.x + threadIdx.x;
    if (e < EE) nrm[e] = -(dinv[dst[e]] * dinv[src[e]]);
}

// Chebyshev coefficients: coe[i] = 2/(K+1) * sum_j relu(temp_j) * T_i(x_j)
__global__ void coeffs_kernel(const float* __restrict__ temp, float* __restrict__ coe) {
    if (threadIdx.x != 0 || blockIdx.x != 0) return;
    const float PI = 3.14159265358979323846f;
    float xj[KORD + 1], rt[KORD + 1], Tp[KORD + 1], Tc[KORD + 1];
    float s0 = 0.0f, s1 = 0.0f;
    for (int j = 0; j <= KORD; ++j) {
        xj[j] = cosf((KORD - j + 0.5f) * PI / (KORD + 1));
        rt[j] = fmaxf(temp[j], 0.0f);
        Tp[j] = 1.0f;
        Tc[j] = xj[j];
        s0 += rt[j];
        s1 += rt[j] * xj[j];
    }
    const float sc = 2.0f / (KORD + 1);
    coe[0] = sc * s0;
    coe[1] = sc * s1;
    for (int i = 2; i <= KORD; ++i) {
        float s = 0.0f;
        for (int j = 0; j <= KORD; ++j) {
            float Tn = 2.0f * xj[j] * Tc[j] - Tp[j];
            Tp[j] = Tc[j];
            Tc[j] = Tn;
            s += rt[j] * Tn;
        }
        coe[i] = sc * s;
    }
}

// ---------------------------------------------------------------- fused MLP
// h = relu(x @ W1 + b1) @ W2 + b2, per-wave 16x64 output strips via WMMA f16.
// W1 (512x64 f32, 128 KB) staged into LDS once per block by the Tensor Data
// Mover; W2 staged as f16; h1 reshaped through LDS between the two GEMMs.
__global__ __launch_bounds__(256) void mlp_kernel(const float* __restrict__ x,
                                                  const float* __restrict__ W1,
                                                  const float* __restrict__ b1,
                                                  const float* __restrict__ W2,
                                                  const float* __restrict__ b2,
                                                  float* __restrict__ h) {
    __shared__ float    sW1f[CIN * CH];      // 128 KB, filled by TDM
    __shared__ _Float16 sW2[CH * CH];        // 8 KB
    __shared__ float    sb1[CH];
    __shared__ float    sb2[CH];
    __shared__ _Float16 sH1[8][16 * CH];     // 16 KB: per-wave 16x64 h1 staging

    const int lane = threadIdx.x & 31;
    const int wave = threadIdx.x >> 5;

#if USE_TDM
    if (wave == 0) {
        unsigned long long ga = (unsigned long long)(uintptr_t)W1;
        unsigned int ldsa = (unsigned int)(uintptr_t)(&sW1f[0]);
        v4u g0;
        g0[0] = 1u;                                              // count=1, user desc
        g0[1] = ldsa;                                            // lds_addr
        g0[2] = (unsigned int)(ga & 0xFFFFFFFFu);                // global_addr[31:0]
        g0[3] = (unsigned int)((ga >> 32) & 0x1FFFFFFu)          // global_addr[56:32]
              | (2u << 30);                                      // type=2 ("image")
        v8i_t g1;
        g1[0] = (int)(2u << 16);        // workgroup_mask=0, data_size=2 (4 bytes)
        g1[1] = (int)(64u << 16);       // tensor_dim0[15:0]=64 at bits[63:48]
        g1[2] = (int)(512u << 16);      // tensor_dim1[15:0]=512 at bits[95:80]
        g1[3] = (int)(64u << 16);       // tile_dim0=64 at bits[127:112]
        g1[4] = (int)512;               // tile_dim1=512 at bits[143:128]; tile_dim2=0
        g1[5] = 64;                     // tensor_dim0_stride[31:0]=64
        g1[6] = 0;                      // stride hi / dim1_stride lo (unused, 2D)
        g1[7] = 0;
        v4i_t gz4 = {0, 0, 0, 0};
        v8i_t gz8 = {0, 0, 0, 0, 0, 0, 0, 0};
        __builtin_amdgcn_tensor_load_to_lds(g0, g1, gz4, gz4, gz8, 0);
        __builtin_amdgcn_s_wait_tensorcnt(0);
    }
#else
    for (int i = threadIdx.x; i < CIN * CH; i += 256) sW1f[i] = W1[i];
#endif

    for (int i = threadIdx.x; i < CH * CH; i += 256) sW2[i] = (_Float16)W2[i];
    if (threadIdx.x < CH) {
        sb1[threadIdx.x] = b1[threadIdx.x];
        sb2[threadIdx.x] = b2[threadIdx.x];
    }
    __syncthreads();

    const int m = lane & 15;             // row within tile (A layout), col lane (B/C layout)
    const int g = lane >> 4;             // lane group 0/1
    const int rowBase = blockIdx.x * 128 + wave * 16;

    int rowA = rowBase + m;
    if (rowA >= NN) rowA = NN - 1;       // clamp loads; stores are guarded
    const float* xr = x + (long long)rowA * CIN;

    // ---- GEMM 1: h1 = x @ W1  (16x512 * 512x64), K in chunks of 32
    v8f acc[4] = {};
    for (int k0 = 0; k0 < CIN; k0 += 32) {
        // A fragment (16x32 f16): lane group g holds K g*8..g*8+7 and g*8+16..g*8+23
        const float4* xa = (const float4*)(xr + k0 + g * 8);
        const float4* xb = (const float4*)(xr + k0 + g * 8 + 16);
        float4 p0 = xa[0], p1 = xa[1], p2 = xb[0], p3 = xb[1];
        v16h a;
        a[0]  = (_Float16)p0.x; a[1]  = (_Float16)p0.y; a[2]  = (_Float16)p0.z; a[3]  = (_Float16)p0.w;
        a[4]  = (_Float16)p1.x; a[5]  = (_Float16)p1.y; a[6]  = (_Float16)p1.z; a[7]  = (_Float16)p1.w;
        a[8]  = (_Float16)p2.x; a[9]  = (_Float16)p2.y; a[10] = (_Float16)p2.z; a[11] = (_Float16)p2.w;
        a[12] = (_Float16)p3.x; a[13] = (_Float16)p3.y; a[14] = (_Float16)p3.z; a[15] = (_Float16)p3.w;
#pragma unroll
        for (int t = 0; t < 4; ++t) {
            // B fragment (32x16 f16): lane group g holds K g*8..g*8+7 and g*8+16..g*8+23
            v16h b;
#pragma unroll
            for (int e = 0; e < 16; ++e) {
                int kk = k0 + g * 8 + (e & 7) + (e >> 3) * 16;
                b[e] = (_Float16)sW1f[kk * CH + t * 16 + m];
            }
            acc[t] = __builtin_amdgcn_wmma_f32_16x16x32_f16(
                false, a, false, b, (short)0, acc[t], false, false);
        }
    }

    // bias + relu, reshape C layout -> row-major f16 in LDS for GEMM 2's A operand
#pragma unroll
    for (int t = 0; t < 4; ++t) {
#pragma unroll
        for (int j = 0; j < 8; ++j) {
            float v = acc[t][j] + sb1[t * 16 + m];
            v = fmaxf(v, 0.0f);
            sH1[wave][(j + g * 8) * CH + t * 16 + m] = (_Float16)v;
        }
    }
    __syncthreads();

    // ---- GEMM 2: h = h1 @ W2  (16x64 * 64x64)
    v8f d[4] = {};
#pragma unroll
    for (int k0 = 0; k0 < CH; k0 += 32) {
        v16h a;
#pragma unroll
        for (int e = 0; e < 16; ++e) {
            int kk = k0 + g * 8 + (e & 7) + (e >> 3) * 16;
            a[e] = sH1[wave][m * CH + kk];
        }
#pragma unroll
        for (int t = 0; t < 4; ++t) {
            v16h b;
#pragma unroll
            for (int e = 0; e < 16; ++e) {
                int kk = k0 + g * 8 + (e & 7) + (e >> 3) * 16;
                b[e] = sW2[kk * CH + t * 16 + m];
            }
            d[t] = __builtin_amdgcn_wmma_f32_16x16x32_f16(
                false, a, false, b, (short)0, d[t], false, false);
        }
    }

    // store: C/D layout lane l<16 -> M=j, l>=16 -> M=j+8; col = t*16 + (l&15)
#pragma unroll
    for (int t = 0; t < 4; ++t) {
#pragma unroll
        for (int j = 0; j < 8; ++j) {
            int r = rowBase + j + g * 8;
            if (r < NN) h[(long long)r * CH + t * 16 + m] = d[t][j] + sb2[t * 16 + m];
        }
    }
}

// ---------------------------------------------------------------- propagation
// out[dst] += norm[e] * v[src], 16 threads/edge, float4 per thread
__global__ __launch_bounds__(256) void spmv_kernel(const int* __restrict__ src,
                                                   const int* __restrict__ dst,
                                                   const float* __restrict__ nrm,
                                                   const float* __restrict__ v,
                                                   float* __restrict__ out) {
    long long t = (long long)blockIdx.x * blockDim.x + threadIdx.x;
    long long e = t >> 4;
    if (e >= EE) return;
    int f = ((int)t & 15) * 4;
    int s = src[e], dd = dst[e];
    float w = nrm[e];
    const float4 vv = *(const float4*)(v + (long long)s * CH + f);
    float* o = out + (long long)dd * CH + f;
    atomicAdd(o + 0, w * vv.x);
    atomicAdd(o + 1, w * vv.y);
    atomicAdd(o + 2, w * vv.z);
    atomicAdd(o + 3, w * vv.w);
}

__global__ __launch_bounds__(256) void out_init_kernel(const float* __restrict__ Tx0,
                                                       const float* __restrict__ Tx1,
                                                       const float* __restrict__ coe,
                                                       float* __restrict__ out, long long n) {
    long long i = (long long)blockIdx.x * blockDim.x + threadIdx.x;
    if (i < n) out[i] = 0.5f * coe[0] * Tx0[i] + coe[1] * Tx1[i];
}

// acc holds Lx(Tx1); produce Tx2 = 2*acc - Tx0 in place, accumulate into out,
// and zero the consumed Tx0 buffer (it becomes the next scatter target).
__global__ __launch_bounds__(256) void cheb_step_kernel(float* __restrict__ acc,
                                                        float* __restrict__ Tx0,
                                                        const float* __restrict__ coe, int ci,
                                                        float* __restrict__ out, long long n) {
    long long i = (long long)blockIdx.x * blockDim.x + threadIdx.x;
    if (i < n) {
        float t0 = Tx0[i];
        float t2 = 2.0f * acc[i] - t0;
        acc[i] = t2;
        out[i] += coe[ci] * t2;
        Tx0[i] = 0.0f;
    }
}

// ---------------------------------------------------------------- launcher
extern "C" void kernel_launch(void* const* d_in, const int* in_sizes, int n_in,
                              void* d_out, int out_size, void* d_ws, size_t ws_size,
                              hipStream_t stream) {
    const float* x    = (const float*)d_in[0];
    const int*   ei   = (const int*)d_in[1];   // [2, E]: row0=src, row1=dst
    const float* W1   = (const float*)d_in[2];
    const float* b1   = (const float*)d_in[3];
    const float* W2   = (const float*)d_in[4];
    const float* b2   = (const float*)d_in[5];
    const float* temp = (const float*)d_in[6];
    float* out = (float*)d_out;

    const int* src = ei;
    const int* dst = ei + EE;

    float* ws   = (float*)d_ws;
    float* coe  = ws;                      // 16
    float* deg  = ws + 16;                 // NN
    float* dinv = deg + NN;                // NN
    float* nrm  = dinv + NN;               // EE
    float* bufA = nrm + EE;                // NN*CH
    float* bufB = bufA + (long long)NN * CH;
    float* bufC = bufB + (long long)NN * CH;

    const long long nc = (long long)NN * CH;
    const int nBlkN  = (NN + 255) / 256;
    const int nBlkE  = (EE + 255) / 256;
    const int nBlkC  = (int)((nc + 255) / 256);
    const int nBlkSp = (int)(((long long)EE * 16 + 255) / 256);

    // normalization
    zero_f32<<<nBlkN, 256, 0, stream>>>(deg, (long long)NN);
    degree_kernel<<<nBlkE, 256, 0, stream>>>(dst, deg);
    dinv_kernel<<<nBlkN, 256, 0, stream>>>(deg, dinv);
    norm_kernel<<<nBlkE, 256, 0, stream>>>(src, dst, dinv, nrm);
    coeffs_kernel<<<1, 32, 0, stream>>>(temp, coe);

    // MLP -> bufA (= h = Tx0)
    mlp_kernel<<<(NN + 127) / 128, 256, 0, stream>>>(x, W1, b1, W2, b2, bufA);

    // Tx1 = Lx(h)
    zero_f32<<<nBlkC, 256, 0, stream>>>(bufB, nc);
    spmv_kernel<<<nBlkSp, 256, 0, stream>>>(src, dst, nrm, bufA, bufB);
    out_init_kernel<<<nBlkC, 256, 0, stream>>>(bufA, bufB, coe, out, nc);

    // bufC must start zeroed for the first scatter of the recurrence
    zero_f32<<<nBlkC, 256, 0, stream>>>(bufC, nc);

    float* Tx0 = bufA;
    float* Tx1 = bufB;
    float* Sp  = bufC;
    for (int i = 2; i <= KORD; ++i) {
        spmv_kernel<<<nBlkSp, 256, 0, stream>>>(src, dst, nrm, Tx1, Sp);
        // consumes Tx0 and zeroes it; Tx0 becomes the next iteration's scatter buffer
        cheb_step_kernel<<<nBlkC, 256, 0, stream>>>(Sp, Tx0, coe, i, out, nc);
        float* t = Tx0; Tx0 = Tx1; Tx1 = Sp; Sp = t;
    }
}
